// EdgeUpdate_61838939128121
// MI455X (gfx1250) — matrix operand order repaired and checked by reading, not compiled, not run on hardware
//
#include <hip/hip_runtime.h>
#include <hip/hip_bf16.h>

// ---------------------------------------------------------------------------
// Fused GNN edge-update for MI455X (gfx1250, wave32, WMMA).
//   Kernel A: vsk = feat@W_vsk + b, vrk = feat@W_vrk + b   (nodes, -> d_ws)
//   Kernel B: out = relu(relu(relu(vsk[src]+vrk[dst] + efeat@W_ek + b_ek)
//                               @W1 + b1) @W2 + b2)         (edges, fused)
// All matmuls via v_wmma_f32_16x16x32_bf16 (bf16 operands, f32 accum).
// Hot loops are branch-free full tiles; the single partial tail tile is
// handled once, after the loop, by the block that owns it.
// ---------------------------------------------------------------------------

#define IN_F  64
#define HID_F 64
#define OUT_F 32

typedef __attribute__((ext_vector_type(16))) __bf16 v16bf;
typedef __attribute__((ext_vector_type(8)))  float  v8f;

union FragB16 { unsigned int u[8]; v16bf v; };

// A-fragment: 16x32 bf16, row-major source with ldk=64 (ISA 7.12.2 layout):
// lanes 0-15 hold row M=lane, K = kb..kb+7 and kb+16..kb+23 (kb = hi*8).
__device__ __forceinline__ v16bf load_a_frag(const __bf16* base,
                                             int row, int ktile, int hi) {
    FragB16 f;
    const unsigned int* p =
        (const unsigned int*)(base + row * 64 + ktile * 32 + hi * 8);
#pragma unroll
    for (int i = 0; i < 4; ++i) { f.u[i] = p[i]; f.u[i + 4] = p[i + 8]; }
    return f.v;
}

// B-fragment: 32x16 bf16 from a transposed weight image Wt[col][k] (ldk=64):
// lane holds column N=lo; K = kb..kb+7 and kb+16..kb+23 (kb = hi*8), as A.
__device__ __forceinline__ v16bf load_b_frag(const __bf16* wt,
                                             int col, int ktile, int hi) {
    FragB16 f;
    const unsigned int* p =
        (const unsigned int*)(wt + col * 64 + ktile * 32 + hi * 8);
#pragma unroll
    for (int i = 0; i < 4; ++i) { f.u[i] = p[i]; f.u[i + 4] = p[i + 8]; }
    return f.v;
}

__device__ __forceinline__ v8f wmma_bf16(v16bf a, v16bf b, v8f c) {
    return __builtin_amdgcn_wmma_f32_16x16x32_bf16(
        false, a, false, b, (short)0, c, false, false);
}

__device__ __forceinline__ float frelu(float v) { return v > 0.0f ? v : 0.0f; }

// ---------------------------------------------------------------------------
// Kernel A tile body (32 nodes x 64 cols, 8 waves: 2x4 16x16 tiles).
// ---------------------------------------------------------------------------
template <bool GUARD>
__device__ __forceinline__ void node_tile_body(
    int nb, int nNodes,
    const float* __restrict__ feat,
    float* __restrict__ vsk, float* __restrict__ vrk,
    __bf16* sX,
    v16bf bs0, v16bf bs1, v16bf br0, v16bf br1,
    float biasS, float biasR,
    int tid, int lo, int hi, int mT, int col) {
    {   // stage 32x64 feature tile -> bf16 LDS
        const int e = tid >> 3, ko = (tid & 7) * 8;
        int node = nb + e;
        if (GUARD && node >= nNodes) node = nNodes - 1;
        const float* sp = feat + (size_t)node * IN_F + ko;
        __bf16* dp = sX + e * 64 + ko;
#pragma unroll
        for (int i = 0; i < 8; ++i) dp[i] = (__bf16)sp[i];
    }
    __syncthreads();

    const v16bf a0 = load_a_frag(sX, mT * 16 + lo, 0, hi);
    const v16bf a1 = load_a_frag(sX, mT * 16 + lo, 1, hi);
    v8f cs = {}, cr = {};
    cs = wmma_bf16(a0, bs0, cs); cs = wmma_bf16(a1, bs1, cs);
    cr = wmma_bf16(a0, br0, cr); cr = wmma_bf16(a1, br1, cr);

    const int row0 = mT * 16 + hi * 8;
    float* ps = vsk + (size_t)(nb + row0) * HID_F + col;  // base + imm offsets
    float* pr = vrk + (size_t)(nb + row0) * HID_F + col;
#pragma unroll
    for (int r = 0; r < 8; ++r) {
        if (!GUARD || (nb + row0 + r) < nNodes) {
            ps[r * HID_F] = cs[r] + biasS;
            pr[r * HID_F] = cr[r] + biasR;
        }
    }
    __syncthreads();
}

__global__ __launch_bounds__(256)
void node_proj_kernel(const float* __restrict__ feat,
                      const float* __restrict__ Wvs, const float* __restrict__ bvs,
                      const float* __restrict__ Wvr, const float* __restrict__ bvr,
                      float* __restrict__ vsk, float* __restrict__ vrk,
                      int nNodes) {
    __shared__ __align__(16) __bf16 sWvs[64 * 64];  // transposed [n][k]
    __shared__ __align__(16) __bf16 sWvr[64 * 64];
    __shared__ __align__(16) __bf16 sX[32 * 64];
    __shared__ float sbvs[64], sbvr[64];

    const int tid = threadIdx.x;
    for (int i = tid; i < 64 * 64; i += 256) {
        const int k = i >> 6, n = i & 63;
        sWvs[n * 64 + k] = (__bf16)Wvs[i];
        sWvr[n * 64 + k] = (__bf16)Wvr[i];
    }
    if (tid < 64) { sbvs[tid] = bvs[tid]; sbvr[tid] = bvr[tid]; }
    __syncthreads();

    const int wave = tid >> 5, lane = tid & 31;
    const int lo = lane & 15, hi = lane >> 4;
    const int mT = wave >> 2, nT = wave & 3;
    const int col = nT * 16 + lo;

    const v16bf bs0 = load_b_frag(sWvs, col, 0, hi);
    const v16bf bs1 = load_b_frag(sWvs, col, 1, hi);
    const v16bf br0 = load_b_frag(sWvr, col, 0, hi);
    const v16bf br1 = load_b_frag(sWvr, col, 1, hi);
    const float biasS = sbvs[col], biasR = sbvr[col];

    const int fullTiles = nNodes >> 5;            // 32-node tiles
    for (int t = blockIdx.x; t < fullTiles; t += gridDim.x)
        node_tile_body<false>(t * 32, nNodes, feat, vsk, vrk, sX,
                              bs0, bs1, br0, br1, biasS, biasR,
                              tid, lo, hi, mT, col);

    if ((nNodes & 31) &&
        (int)(fullTiles % gridDim.x) == (int)blockIdx.x)  // tail owner
        node_tile_body<true>(fullTiles * 32, nNodes, feat, vsk, vrk, sX,
                             bs0, bs1, br0, br1, biasS, biasR,
                             tid, lo, hi, mT, col);
}

// ---------------------------------------------------------------------------
// Kernel B tile body (64 edges x 3 fused WMMA layers).
// Stages 1-2: 4x4 16x16 tiles (2 chains/wave); stage 3: 4x2 (1 chain/wave).
// ---------------------------------------------------------------------------
template <bool GUARD>
__device__ __forceinline__ void edge_tile_body(
    int eb, int nEdges,
    const float* __restrict__ efeat,
    const int* __restrict__ src, const int* __restrict__ dst,
    const float* __restrict__ vsk, const float* __restrict__ vrk,
    float* __restrict__ out,
    __bf16* sX, __bf16* sH, __bf16* sH2, float* sG,
    v16bf bek0, v16bf bek1, v16bf bw10, v16bf bw11, v16bf bw20, v16bf bw21,
    float biasE, float bias1, float bias2,
    int tid, int lo, int hi, int mBase, int col, int m3, int col2) {
    {   // stage efeat tile (bf16) + gathered node sums (f32) into LDS
        const int e = tid >> 2, ko = (tid & 3) * 16;
        int ee = eb + e;
        if (GUARD && ee >= nEdges) ee = nEdges - 1;
        const int s = src[ee], d = dst[ee];
        const float* xp = efeat + (size_t)ee * IN_F + ko;
        const float* gs = vsk + (size_t)s * HID_F + ko;
        const float* gr = vrk + (size_t)d * HID_F + ko;
        __bf16* xd = sX + e * 64 + ko;
        float*  gd = sG + e * 64 + ko;
#pragma unroll
        for (int i = 0; i < 16; ++i) {
            xd[i] = (__bf16)xp[i];
            gd[i] = gs[i] + gr[i];
        }
    }
    __syncthreads();

    // ---- layer 0: relu(efeat@Wek + gather + b_ek) -----------------------
#pragma unroll
    for (int mm = 0; mm < 2; ++mm) {
        const int mT = mBase + mm;
        const v16bf a0 = load_a_frag(sX, mT * 16 + lo, 0, hi);
        const v16bf a1 = load_a_frag(sX, mT * 16 + lo, 1, hi);
        v8f c = {};
        c = wmma_bf16(a0, bek0, c);
        c = wmma_bf16(a1, bek1, c);
        const int base = (mT * 16 + hi * 8) * 64 + col;
#pragma unroll
        for (int r = 0; r < 8; ++r)
            sH[base + r * 64] = (__bf16)frelu(c[r] + sG[base + r * 64] + biasE);
    }
    __syncthreads();

    // ---- layer 1: relu(H@W1 + b1) ---------------------------------------
#pragma unroll
    for (int mm = 0; mm < 2; ++mm) {
        const int mT = mBase + mm;
        const v16bf a0 = load_a_frag(sH, mT * 16 + lo, 0, hi);
        const v16bf a1 = load_a_frag(sH, mT * 16 + lo, 1, hi);
        v8f c = {};
        c = wmma_bf16(a0, bw10, c);
        c = wmma_bf16(a1, bw11, c);
        const int base = (mT * 16 + hi * 8) * 64 + col;
#pragma unroll
        for (int r = 0; r < 8; ++r)
            sH2[base + r * 64] = (__bf16)frelu(c[r] + bias1);
    }
    __syncthreads();

    // ---- layer 2: relu(H2@W2 + b2) -> global ----------------------------
    {
        const v16bf a0 = load_a_frag(sH2, m3 * 16 + lo, 0, hi);
        const v16bf a1 = load_a_frag(sH2, m3 * 16 + lo, 1, hi);
        v8f c = {};
        c = wmma_bf16(a0, bw20, c);
        c = wmma_bf16(a1, bw21, c);
        const int row0 = m3 * 16 + hi * 8;
        float* op = out + (size_t)(eb + row0) * OUT_F + col2;  // base+imm offs
#pragma unroll
        for (int r = 0; r < 8; ++r) {
            if (!GUARD || (eb + row0 + r) < nEdges)
                op[r * OUT_F] = frelu(c[r] + bias2);
        }
    }
    __syncthreads();
}

__global__ __launch_bounds__(256)
void edge_fused_kernel(const float* __restrict__ efeat,
                       const int* __restrict__ src, const int* __restrict__ dst,
                       const float* __restrict__ vsk, const float* __restrict__ vrk,
                       const float* __restrict__ Wek, const float* __restrict__ bek,
                       const float* __restrict__ W1,  const float* __restrict__ b1,
                       const float* __restrict__ W2,  const float* __restrict__ b2,
                       float* __restrict__ out, int nEdges) {
    __shared__ __align__(16) __bf16 sWek[64 * 64];  // transposed
    __shared__ __align__(16) __bf16 sW1[64 * 64];   // transposed
    __shared__ __align__(16) __bf16 sW2[32 * 64];   // transposed
    __shared__ __align__(16) __bf16 sX[64 * 64];    // efeat tile (bf16)
    __shared__ __align__(16) __bf16 sH[64 * 64];    // layer-0 act
    __shared__ __align__(16) __bf16 sH2[64 * 64];   // layer-1 act
    __shared__ float sG[64 * 64];                   // vsk[src]+vrk[dst]
    __shared__ float sbek[64], sb1[64], sb2[32];

    const int tid = threadIdx.x;
    for (int i = tid; i < 64 * 64; i += 256) {
        const int k = i >> 6, n = i & 63;
        sWek[n * 64 + k] = (__bf16)Wek[i];
        sW1[n * 64 + k]  = (__bf16)W1[i];
    }
    for (int i = tid; i < 64 * 32; i += 256) {
        const int k = i >> 5, n = i & 31;               // W2 is [64][32]
        sW2[n * 64 + k] = (__bf16)W2[i];
    }
    if (tid < 64) { sbek[tid] = bek[tid]; sb1[tid] = b1[tid]; }
    if (tid < 32) { sb2[tid] = b2[tid]; }
    __syncthreads();

    const int wave = tid >> 5, lane = tid & 31;
    const int lo = lane & 15, hi = lane >> 4;
    const int nT = wave & 3;                   // stages 1-2: 4 n-tiles
    const int mBase = (wave >> 2) * 2;         // stages 1-2: 2 m-tiles/wave
    const int col = nT * 16 + lo;
    const int m3 = wave >> 1;                  // stage 3: 4 m-tiles x 2 n-tiles
    const int col2 = (wave & 1) * 16 + lo;

    const v16bf bek0 = load_b_frag(sWek, col, 0, hi);
    const v16bf bek1 = load_b_frag(sWek, col, 1, hi);
    const v16bf bw10 = load_b_frag(sW1, col, 0, hi);
    const v16bf bw11 = load_b_frag(sW1, col, 1, hi);
    const v16bf bw20 = load_b_frag(sW2, col2, 0, hi);
    const v16bf bw21 = load_b_frag(sW2, col2, 1, hi);
    const float biasE = sbek[col], bias1 = sb1[col], bias2 = sb2[col2];

    const int fullTiles = nEdges >> 6;            // 64-edge tiles
    for (int t = blockIdx.x; t < fullTiles; t += gridDim.x)
        edge_tile_body<false>(t * 64, nEdges, efeat, src, dst, vsk, vrk, out,
                              sX, sH, sH2, sG,
                              bek0, bek1, bw10, bw11, bw20, bw21,
                              biasE, bias1, bias2,
                              tid, lo, hi, mBase, col, m3, col2);

    if ((nEdges & 63) &&
        (int)(fullTiles % gridDim.x) == (int)blockIdx.x)  // tail owner
        edge_tile_body<true>(fullTiles * 64, nEdges, efeat, src, dst, vsk, vrk,
                             out, sX, sH, sH2, sG,
                             bek0, bek1, bw10, bw11, bw20, bw21,
                             biasE, bias1, bias2,
                             tid, lo, hi, mBase, col, m3, col2);
}

// ---------------------------------------------------------------------------
extern "C" void kernel_launch(void* const* d_in, const int* in_sizes, int n_in,
                              void* d_out, int out_size, void* d_ws, size_t ws_size,
                              hipStream_t stream) {
    const float* feat  = (const float*)d_in[0];
    const float* efeat = (const float*)d_in[1];
    const int*   src   = (const int*)d_in[2];
    const int*   dst   = (const int*)d_in[3];
    const float* Wvs   = (const float*)d_in[4];
    const float* bvs   = (const float*)d_in[5];
    const float* Wvr   = (const float*)d_in[6];
    const float* bvr   = (const float*)d_in[7];
    const float* Wek   = (const float*)d_in[8];
    const float* bek   = (const float*)d_in[9];
    const float* W1    = (const float*)d_in[10];
    const float* b1    = (const float*)d_in[11];
    const float* W2    = (const float*)d_in[12];
    const float* b2    = (const float*)d_in[13];
    float* out = (float*)d_out;

    const int N = in_sizes[0] / IN_F;
    const int E = in_sizes[2];

    float* vsk = (float*)d_ws;
    float* vrk = vsk + (size_t)N * HID_F;

    const int tA = (N + 31) / 32;
    const int tB = (E + 63) / 64;
    const int gA = tA < 1024 ? tA : 1024;
    const int gB = tB < 4096 ? tB : 4096;

    node_proj_kernel<<<dim3(gA), dim3(256), 0, stream>>>(
        feat, Wvs, bvs, Wvr, bvr, vsk, vrk, N);
    edge_fused_kernel<<<dim3(gB), dim3(256), 0, stream>>>(
        efeat, src, dst, vsk, vrk, Wek, bek, W1, b1, W2, b2, out, E);
}
